// MinimalMOIRAI_33285996544532
// MI455X (gfx1250) — matrix operand order repaired and checked by reading, not compiled, not run on hardware
//
#include <hip/hip_runtime.h>
#include <hip/hip_bf16.h>
#include <math.h>

// ---------------------------------------------------------------------------
// MI455X (gfx1250, wave32) bf16-WMMA implementation of the MOIRAI reference.
// Heavy matmuls run on v_wmma_f32_16x16x32_bf16; global->LDS staging uses the
// CDNA5 async-to-LDS path (ASYNCcnt) with double-buffered tiles so the DMA of
// tile t+1 overlaps the WMMA consumption of tile t.
// ---------------------------------------------------------------------------

typedef __bf16 bf16;
typedef bf16  v8bf  __attribute__((ext_vector_type(8)));
typedef bf16  v16bf __attribute__((ext_vector_type(16)));
typedef float v8f   __attribute__((ext_vector_type(8)));

#define WMMA_BF16(A_, B_, C_) \
  __builtin_amdgcn_wmma_f32_16x16x32_bf16(false, (A_), false, (B_), (short)0, (C_), false, false)

static __device__ __forceinline__ v16bf cat8(v8bf lo, v8bf hi8) {
  return __builtin_shufflevector(lo, hi8, 0,1,2,3,4,5,6,7,8,9,10,11,12,13,14,15);
}

// low 32 bits of a generic pointer to an LDS object = LDS byte offset
static __device__ __forceinline__ unsigned lds_off(const void* p) {
  return (unsigned)(unsigned long long)p;
}
// CDNA5 async DMA: per-lane 16B global -> LDS, tracked on ASYNCcnt
static __device__ __forceinline__ void async_ld_b128(unsigned lds_addr, const void* gptr) {
  asm volatile("global_load_async_to_lds_b128 %0, %1, off"
               :: "v"(lds_addr), "v"((unsigned long long)(unsigned long)gptr)
               : "memory");
}
static __device__ __forceinline__ void wait_async0() {
  asm volatile("s_wait_asynccnt 0" ::: "memory");
}

constexpr int kB   = 4;
constexpr int kS   = 2048;
constexpr int kP   = 32;
constexpr int kD   = 512;
constexpr int kH   = 8;
constexpr int kHD  = 64;
constexpr int kDFF = 2048;
constexpr int kM   = kB * kS;   // 8192 tokens

// ---------------------------------------------------------------------------
// Generic GEMM: C[M,N] = A[M,K](bf16,row-major) x Bt[N,K](bf16,row-major)^T
// + bias, optional exact GELU. Writes f32 and/or bf16 outputs.
// Block = 256 thr (8 waves). Block tile 128x128; wave tile 32x64 (2x4 WMMA).
// Double-buffered async-to-LDS staging of A and Bt k-tiles.
// Requires M%128==0, N%128==0, K%32==0.
// ---------------------------------------------------------------------------
__global__ __launch_bounds__(256) void wmma_gemm_kernel(
    const bf16* __restrict__ A, const bf16* __restrict__ Bt,
    const float* __restrict__ bias, float* __restrict__ outF,
    bf16* __restrict__ outB, int Nd, int Kd, int gelu)
{
  __shared__ bf16 As[2][128 * 32];
  __shared__ bf16 Bs[2][128 * 32];

  const int tid  = threadIdx.x;
  const int lane = tid & 31;
  const int wave = tid >> 5;
  const int l15  = lane & 15;
  const int hi   = lane >> 4;          // which 16-lane half
  const int wm   = wave & 3;           // wave row 0..3 (32 rows each)
  const int wn   = wave >> 2;          // wave col 0..1 (64 cols each)
  const int m0   = blockIdx.y * 128;
  const int n0   = blockIdx.x * 128;

  const int r0 = tid >> 2;             // staging: row 0..63 (+64 for 2nd half)
  const int p0 = tid & 3;              // staging: 16B chunk within 64B row

  v8f acc[2][4] = {};

  const int T = Kd >> 5;               // number of 32-wide k tiles

  // issue async DMA of k-tile kt into buffer kt&1 (4 x b128 per thread)
  auto issue = [&](int kt) {
    int k0 = kt << 5, bsel = kt & 1;
    async_ld_b128(lds_off(&As[bsel][r0 * 32 + p0 * 8]),
                  &A[(size_t)(m0 + r0) * Kd + k0 + p0 * 8]);
    async_ld_b128(lds_off(&As[bsel][(r0 + 64) * 32 + p0 * 8]),
                  &A[(size_t)(m0 + r0 + 64) * Kd + k0 + p0 * 8]);
    async_ld_b128(lds_off(&Bs[bsel][r0 * 32 + p0 * 8]),
                  &Bt[(size_t)(n0 + r0) * Kd + k0 + p0 * 8]);
    async_ld_b128(lds_off(&Bs[bsel][(r0 + 64) * 32 + p0 * 8]),
                  &Bt[(size_t)(n0 + r0 + 64) * Kd + k0 + p0 * 8]);
  };

  issue(0);
  for (int kt = 0; kt < T; kt++) {
    wait_async0();                      // my DMA for tile kt is done
    __syncthreads();                    // everyone's DMA visible, prev reads done
    if (kt + 1 < T) issue(kt + 1);      // prefetch overlaps compute below
    const bf16* Ab = As[kt & 1];
    const bf16* Bb = Bs[kt & 1];

    v16bf af[2], bfv[4];
    #pragma unroll
    for (int mi = 0; mi < 2; mi++) {
      int row = wm * 32 + mi * 16 + l15;
      int kb  = hi * 8;                 // A layout: 0..7/16..23 | 8..15/24..31
      af[mi] = cat8(*(const v8bf*)&Ab[row * 32 + kb],
                    *(const v8bf*)&Ab[row * 32 + kb + 16]);
    }
    #pragma unroll
    for (int ni = 0; ni < 4; ni++) {
      int col = wn * 64 + ni * 16 + l15;
      int kb  = hi * 8;                 // B layout: 0..7/16..23 | 8..15/24..31
      bfv[ni] = cat8(*(const v8bf*)&Bb[col * 32 + kb],
                     *(const v8bf*)&Bb[col * 32 + kb + 16]);
    }
    #pragma unroll
    for (int mi = 0; mi < 2; mi++)
      #pragma unroll
      for (int ni = 0; ni < 4; ni++)
        acc[mi][ni] = WMMA_BF16(af[mi], bfv[ni], acc[mi][ni]);
  }

  // epilogue: C/D layout -> row = r + 8*hi, col = l15 within each 16x16 tile
  #pragma unroll
  for (int mi = 0; mi < 2; mi++) {
    #pragma unroll
    for (int ni = 0; ni < 4; ni++) {
      int col = n0 + wn * 64 + ni * 16 + l15;
      float bv = bias ? bias[col] : 0.f;
      #pragma unroll
      for (int r = 0; r < 8; r++) {
        int row = m0 + wm * 32 + mi * 16 + r + hi * 8;
        float v = acc[mi][ni][r] + bv;
        if (gelu) v = 0.5f * v * (1.f + erff(v * 0.70710678f));
        size_t o = (size_t)row * Nd + col;
        if (outF) outF[o] = v;
        if (outB) outB[o] = (bf16)v;
      }
    }
  }
}

// ---------------------------------------------------------------------------
// Flash-style attention. 8 waves per block share one (b,h); each wave owns a
// 16-query tile. K (32x64) and Vt (64x32) key-block tiles are staged once per
// block via double-buffered async-to-LDS DMA. qbf/kbf: [b,h,s,hd] bf16;
// vtbf: [b,h,hd,s] bf16. Output attnbf: [b,s,h*hd] bf16.
// ---------------------------------------------------------------------------
__global__ __launch_bounds__(256) void attn_kernel(
    const bf16* __restrict__ qbf, const bf16* __restrict__ kbf,
    const bf16* __restrict__ vtbf, const int* __restrict__ vids,
    const float* __restrict__ mask, const float* __restrict__ u_same,
    const float* __restrict__ u_cross, bf16* __restrict__ outbf)
{
  __shared__ bf16 Ks[2][32 * 64];       // [buf][key][feat]
  __shared__ bf16 Vs[2][64 * 32];       // [buf][hd][key]
  __shared__ bf16 Pbuf[8 * 16 * 32];    // per-wave P staging (A-frag relayout)

  const int tid  = threadIdx.x;
  const int lane = tid & 31;
  const int wave = tid >> 5;
  const int l15  = lane & 15;
  const int hi   = lane >> 4;
  bf16* Pw = &Pbuf[wave * 16 * 32];

  const int bh = blockIdx.x >> 4;       // 16 blocks per (b,h)
  const int hh = bh & 7, bb = bh >> 3;
  const int q0 = ((blockIdx.x & 15) * 8 + wave) * 16;

  const size_t base_qk = (size_t)bh * kS * kHD;
  const size_t base_vt = (size_t)bh * kHD * kS;
  const float us = u_same[hh], uc = u_cross[hh];
  const float scale = 0.125f;           // 1/sqrt(64)

  // staging indices (per thread, one b128 chunk each for K and Vt)
  const int kr = tid >> 3, kp = tid & 7;   // K:  key 0..31, 16B chunk 0..7
  const int vr = tid >> 2, vp = tid & 3;   // Vt: hd  0..63, 16B chunk 0..3

  auto issue = [&](int kb) {
    int key0 = kb << 5, bsel = kb & 1;
    async_ld_b128(lds_off(&Ks[bsel][kr * 64 + kp * 8]),
                  &kbf[base_qk + (size_t)(key0 + kr) * kHD + kp * 8]);
    async_ld_b128(lds_off(&Vs[bsel][vr * 32 + vp * 8]),
                  &vtbf[base_vt + (size_t)vr * kS + key0 + vp * 8]);
  };

  // Q fragments (HD=64 -> two 32-wide k-steps), loaded once
  v16bf qf[2];
  #pragma unroll
  for (int ks = 0; ks < 2; ks++) {
    const bf16* qp = qbf + base_qk + (size_t)(q0 + l15) * kHD + ks * 32 + hi * 8;
    qf[ks] = cat8(*(const v8bf*)qp, *(const v8bf*)(qp + 16));
  }

  int vq[8];
  #pragma unroll
  for (int r = 0; r < 8; r++) vq[r] = vids[bb * kS + q0 + r + hi * 8];

  float mrow[8], lrow[8];
  v8f oacc[4] = {};
  #pragma unroll
  for (int r = 0; r < 8; r++) { mrow[r] = -1e30f; lrow[r] = 0.f; }

  issue(0);
  for (int kb = 0; kb < kS / 32; kb++) {
    wait_async0();
    __syncthreads();
    if (kb + 1 < kS / 32) issue(kb + 1);
    const bf16* Kb = Ks[kb & 1];
    const bf16* Vb = Vs[kb & 1];
    int key0 = kb * 32;

    // scores: 16 x 32 = two 16x16 WMMA tiles, each accumulated over HD
    v8f sacc[2] = {};
    #pragma unroll
    for (int nk = 0; nk < 2; nk++) {
      int kloc = nk * 16 + l15;
      #pragma unroll
      for (int ks = 0; ks < 2; ks++) {
        const bf16* kp2 = &Kb[kloc * 64 + ks * 32 + hi * 8];
        v16bf kf = cat8(*(const v8bf*)kp2, *(const v8bf*)(kp2 + 16));
        sacc[nk] = WMMA_BF16(qf[ks], kf, sacc[nk]);
      }
    }

    // bias + mask, then online softmax update
    float sv[2][8];
    #pragma unroll
    for (int nk = 0; nk < 2; nk++) {
      int keyn = key0 + nk * 16 + l15;
      int vk = vids[bb * kS + keyn];
      float ma = (1.f - mask[bb * kS + keyn]) * -1e9f;
      #pragma unroll
      for (int r = 0; r < 8; r++)
        sv[nk][r] = sacc[nk][r] * scale + ((vq[r] == vk) ? us : uc) + ma;
    }
    float fac[8];
    #pragma unroll
    for (int r = 0; r < 8; r++) {
      float t = fmaxf(sv[0][r], sv[1][r]);
      #pragma unroll
      for (int off = 1; off < 16; off <<= 1) t = fmaxf(t, __shfl_xor(t, off, 16));
      float mn = fmaxf(mrow[r], t);
      fac[r] = __expf(mrow[r] - mn);
      mrow[r] = mn;
      float p0 = __expf(sv[0][r] - mn);
      float p1 = __expf(sv[1][r] - mn);
      sv[0][r] = p0; sv[1][r] = p1;
      float rs = p0 + p1;
      #pragma unroll
      for (int off = 1; off < 16; off <<= 1) rs += __shfl_xor(rs, off, 16);
      lrow[r] = lrow[r] * fac[r] + rs;
    }
    #pragma unroll
    for (int ni = 0; ni < 4; ni++)
      #pragma unroll
      for (int r = 0; r < 8; r++) oacc[ni][r] *= fac[r];

    // P (C/D layout) -> per-wave LDS -> A-fragment layout
    #pragma unroll
    for (int nk = 0; nk < 2; nk++)
      #pragma unroll
      for (int r = 0; r < 8; r++)
        Pw[(r + hi * 8) * 32 + nk * 16 + l15] = (bf16)sv[nk][r];
    asm volatile("s_wait_dscnt 0" ::: "memory");
    const bf16* pp = &Pw[l15 * 32 + hi * 8];
    v16bf pf = cat8(*(const v8bf*)pp, *(const v8bf*)(pp + 16));

    // O += P x V  (Vt staged in LDS -> contiguous key reads)
    #pragma unroll
    for (int ni = 0; ni < 4; ni++) {
      const bf16* vp2 = &Vb[(ni * 16 + l15) * 32 + hi * 8];
      v16bf vf = cat8(*(const v8bf*)vp2, *(const v8bf*)(vp2 + 16));
      oacc[ni] = WMMA_BF16(pf, vf, oacc[ni]);
    }
  }

  #pragma unroll
  for (int r = 0; r < 8; r++) {
    float inv = 1.f / lrow[r];
    int row = q0 + r + hi * 8;
    #pragma unroll
    for (int ni = 0; ni < 4; ni++) {
      float v = oacc[ni][r] * inv;
      outbf[((size_t)bb * kS + row) * kD + hh * kHD + ni * 16 + l15] = (bf16)v;
    }
  }
}

// ---------------------------------------------------------------------------
// Small helper kernels
// ---------------------------------------------------------------------------
__global__ void pack_wt_kernel(const float* __restrict__ W, bf16* __restrict__ Wt,
                               int K, int N) {
  int idx = blockIdx.x * 256 + threadIdx.x;
  if (idx >= K * N) return;
  int n = idx % N, k = idx / N;
  Wt[(size_t)n * K + k] = (bf16)W[idx];
}

__global__ void embed_kernel(const float* __restrict__ x, const float* __restrict__ W,
                             const float* __restrict__ b, float* __restrict__ hF,
                             bf16* __restrict__ hB) {
  int idx = blockIdx.x * 256 + threadIdx.x;   // over M*D
  int d = idx & (kD - 1); int m = idx >> 9;
  float a = b[d];
  #pragma unroll
  for (int p = 0; p < kP; p++) a += x[m * kP + p] * W[p * kD + d];
  hF[idx] = a; hB[idx] = (bf16)a;
}

__global__ void rope_pack_kernel(const float* __restrict__ t, bf16* __restrict__ out) {
  int idx = blockIdx.x * 256 + threadIdx.x;   // over M*H*32 rotation pairs
  int i = idx & 31; int hh = (idx >> 5) & 7; int m = idx >> 8;
  int s = m & (kS - 1); int b = m >> 11;
  float f = __powf(10000.f, -((float)(i & 15)) / 16.f);
  float sn, cs; __sincosf((float)s * f, &sn, &cs);
  size_t ib = (size_t)m * kD + hh * kHD + 2 * i;
  float x0 = t[ib], x1 = t[ib + 1];
  size_t ob = ((size_t)(b * kH + hh) * kS + s) * kHD + 2 * i;
  out[ob]     = (bf16)(x0 * cs - x1 * sn);
  out[ob + 1] = (bf16)(x0 * sn + x1 * cs);
}

__global__ void vpack_kernel(const float* __restrict__ t, bf16* __restrict__ vt) {
  int idx = blockIdx.x * 256 + threadIdx.x;   // over M*D
  int d = idx & (kD - 1); int m = idx >> 9;
  int hh = d >> 6, hd = d & 63;
  int s = m & (kS - 1), b = m >> 11;
  vt[((size_t)(b * kH + hh) * kHD + hd) * kS + s] = (bf16)t[idx];
}

__global__ __launch_bounds__(256) void add_ln_kernel(
    const float* __restrict__ x, const float* __restrict__ y,
    const float* __restrict__ g, const float* __restrict__ bta,
    float* __restrict__ outF, bf16* __restrict__ outB) {
  __shared__ float red[16];
  int m = blockIdx.x, t = threadIdx.x;
  int lane = t & 31, wave = t >> 5;
  size_t base = (size_t)m * kD;
  float v0 = x[base + t]       + y[base + t];
  float v1 = x[base + t + 256] + y[base + t + 256];
  float s = v0 + v1, sq = v0 * v0 + v1 * v1;
  #pragma unroll
  for (int off = 16; off >= 1; off >>= 1) {
    s  += __shfl_xor(s,  off, 32);
    sq += __shfl_xor(sq, off, 32);
  }
  if (lane == 0) { red[wave] = s; red[8 + wave] = sq; }
  __syncthreads();
  float S1 = 0, S2 = 0;
  #pragma unroll
  for (int i = 0; i < 8; i++) { S1 += red[i]; S2 += red[8 + i]; }
  float mean = S1 * (1.f / 512.f);
  float var  = S2 * (1.f / 512.f) - mean * mean;
  float rstd = rsqrtf(var + 1e-5f);
  float o0 = (v0 - mean) * rstd * g[t]       + bta[t];
  float o1 = (v1 - mean) * rstd * g[t + 256] + bta[t + 256];
  outF[base + t]       = o0; outB[base + t]       = (bf16)o0;
  outF[base + t + 256] = o1; outB[base + t + 256] = (bf16)o1;
}

__global__ void head_kernel(const float* __restrict__ h, const float* __restrict__ Wh,
                            const float* __restrict__ bh, float* __restrict__ out) {
  int idx = blockIdx.x * 256 + threadIdx.x;   // over M*P
  int p = idx & 31; int m = idx >> 5;
  float a = bh[p];
  for (int d = 0; d < kD; d++) a += h[(size_t)m * kD + d] * Wh[d * kP + p];
  out[idx] = a;
}

// ---------------------------------------------------------------------------
extern "C" void kernel_launch(void* const* d_in, const int* in_sizes, int n_in,
                              void* d_out, int out_size, void* d_ws, size_t ws_size,
                              hipStream_t stream) {
  (void)in_sizes; (void)n_in; (void)out_size; (void)ws_size;
  const float* x     = (const float*)d_in[0];
  const int*   vids  = (const int*)  d_in[1];
  const float* mask  = (const float*)d_in[2];
  const float* W_pe  = (const float*)d_in[3];
  const float* b_pe  = (const float*)d_in[4];
  const float* Wq    = (const float*)d_in[5];
  const float* bq    = (const float*)d_in[6];
  const float* Wk    = (const float*)d_in[7];
  const float* bk    = (const float*)d_in[8];
  const float* Wv    = (const float*)d_in[9];
  const float* bv    = (const float*)d_in[10];
  const float* Wo    = (const float*)d_in[11];
  const float* bo    = (const float*)d_in[12];
  const float* u_s   = (const float*)d_in[13];
  const float* u_c   = (const float*)d_in[14];
  const float* g1    = (const float*)d_in[15];
  const float* be1   = (const float*)d_in[16];
  const float* W1    = (const float*)d_in[17];
  const float* b1f   = (const float*)d_in[18];
  const float* W2    = (const float*)d_in[19];
  const float* b2f   = (const float*)d_in[20];
  const float* g2    = (const float*)d_in[21];
  const float* be2   = (const float*)d_in[22];
  const float* Wh    = (const float*)d_in[23];
  const float* bh    = (const float*)d_in[24];
  float* out = (float*)d_out;

  char* ws = (char*)d_ws; size_t off = 0;
  auto alloc = [&](size_t bytes) -> void* {
    void* p = ws + off; off += (bytes + 255) & ~(size_t)255; return p;
  };
  float* h      = (float*)alloc((size_t)kM * kD * 4);
  bf16*  hbf    = (bf16*) alloc((size_t)kM * kD * 2);
  float* tmp    = (float*)alloc((size_t)kM * kD * 4);
  bf16*  qbf    = (bf16*) alloc((size_t)kM * kD * 2);
  bf16*  kbf    = (bf16*) alloc((size_t)kM * kD * 2);
  bf16*  vtbf   = (bf16*) alloc((size_t)kM * kD * 2);
  bf16*  attnbf = (bf16*) alloc((size_t)kM * kD * 2);
  bf16*  f1bf   = (bf16*) alloc((size_t)kM * kDFF * 2);
  bf16*  wt     = (bf16*) alloc((size_t)kDFF * kD * 2);

  embed_kernel<<<kM * kD / 256, 256, 0, stream>>>(x, W_pe, b_pe, h, hbf);

  for (int l = 0; l < 2; l++) {
    const size_t wdd = (size_t)l * kD * kD;
    // Q projection + RoPE
    pack_wt_kernel<<<kD * kD / 256, 256, 0, stream>>>(Wq + wdd, wt, kD, kD);
    wmma_gemm_kernel<<<dim3(kD / 128, kM / 128), 256, 0, stream>>>(
        hbf, wt, bq + l * kD, tmp, nullptr, kD, kD, 0);
    rope_pack_kernel<<<kM * kH * 32 / 256, 256, 0, stream>>>(tmp, qbf);
    // K projection + RoPE
    pack_wt_kernel<<<kD * kD / 256, 256, 0, stream>>>(Wk + wdd, wt, kD, kD);
    wmma_gemm_kernel<<<dim3(kD / 128, kM / 128), 256, 0, stream>>>(
        hbf, wt, bk + l * kD, tmp, nullptr, kD, kD, 0);
    rope_pack_kernel<<<kM * kH * 32 / 256, 256, 0, stream>>>(tmp, kbf);
    // V projection + transpose-pack
    pack_wt_kernel<<<kD * kD / 256, 256, 0, stream>>>(Wv + wdd, wt, kD, kD);
    wmma_gemm_kernel<<<dim3(kD / 128, kM / 128), 256, 0, stream>>>(
        hbf, wt, bv + l * kD, tmp, nullptr, kD, kD, 0);
    vpack_kernel<<<kM * kD / 256, 256, 0, stream>>>(tmp, vtbf);
    // attention
    attn_kernel<<<kB * kH * (kS / 16) / 8, 256, 0, stream>>>(
        qbf, kbf, vtbf, vids, mask, u_s + l * kH, u_c + l * kH, attnbf);
    // O projection + residual LN
    pack_wt_kernel<<<kD * kD / 256, 256, 0, stream>>>(Wo + wdd, wt, kD, kD);
    wmma_gemm_kernel<<<dim3(kD / 128, kM / 128), 256, 0, stream>>>(
        attnbf, wt, bo + l * kD, tmp, nullptr, kD, kD, 0);
    add_ln_kernel<<<kM, 256, 0, stream>>>(h, tmp, g1 + l * kD, be1 + l * kD, h, hbf);
    // FFN
    pack_wt_kernel<<<kD * kDFF / 256, 256, 0, stream>>>(W1 + (size_t)l * kD * kDFF, wt, kD, kDFF);
    wmma_gemm_kernel<<<dim3(kDFF / 128, kM / 128), 256, 0, stream>>>(
        hbf, wt, b1f + l * kDFF, nullptr, f1bf, kDFF, kD, 1);
    pack_wt_kernel<<<kDFF * kD / 256, 256, 0, stream>>>(W2 + (size_t)l * kDFF * kD, wt, kDFF, kD);
    wmma_gemm_kernel<<<dim3(kD / 128, kM / 128), 256, 0, stream>>>(
        f1bf, wt, b2f + l * kD, tmp, nullptr, kD, kDFF, 0);
    add_ln_kernel<<<kM, 256, 0, stream>>>(h, tmp, g2 + l * kD, be2 + l * kD, h, hbf);
  }

  head_kernel<<<kM * kP / 256, 256, 0, stream>>>(h, Wh, bh, out);
}
